// Attention_14757507629874
// MI455X (gfx1250) — matrix operand order repaired and checked by reading, not compiled, not run on hardware
//
#include <hip/hip_runtime.h>
#include <hip/hip_bf16.h>
#include <math.h>

// ---------------------------------------------------------------------------
// MI455X (gfx1250) fused GQA attention block.
//   B=2, T=2048, D=2048, DH=128, NQ=16, NKV=4
// Pipeline:
//   1) qbuf  = x @ Wq      (bf16 WMMA GEMM, async global->LDS A-tiles)
//   2) kvbuf = x @ Wkv     (same GEMM kernel)
//   3) RMSNorm + YaRN RoPE on q / k heads
//   4) causal flash attention (WMMA scores + WMMA P@V, bf16 LDS staging)
//   5) out   = attn @ Wout (same GEMM kernel)
// Workspace: qbuf 32MB + kvbuf 16MB + attn 32MB = 80MB in d_ws.
// ---------------------------------------------------------------------------

typedef __attribute__((ext_vector_type(16))) __bf16 v16bf;
typedef __attribute__((ext_vector_type(8)))  __bf16 v8bf;
typedef __attribute__((ext_vector_type(8)))  float  v8f;
typedef int v4i_async __attribute__((vector_size(16)));   // matches builtin V4i

#define Tt   2048
#define Dd   2048
#define DH   128
#define NQ   16
#define NKV  4

#define ATTN_SCALE (1.0f / ((0.1f * 0.69314718056f + 1.0f) * 11.313708499f))
#define RMS_EPS    1.1920929e-07f

static __device__ __forceinline__ __bf16 tobf(float f) { return (__bf16)f; }

// ---- CDNA5 async global->LDS (ASYNCcnt) with sync fallback -----------------
#if __has_builtin(__builtin_amdgcn_global_load_async_to_lds_b128) && \
    __has_builtin(__builtin_amdgcn_s_wait_asynccnt)
#define HAVE_ASYNC_LDS 1
#endif

static __device__ __forceinline__ void copy16_g2l(const float* g, float* l) {
#ifdef HAVE_ASYNC_LDS
    __builtin_amdgcn_global_load_async_to_lds_b128(
        (__attribute__((address_space(1))) v4i_async*)g,
        (__attribute__((address_space(3))) v4i_async*)l, 0, 0);
#else
    *(float4*)l = *(const float4*)g;
#endif
}

template <int N> static __device__ __forceinline__ void wait_async() {
#ifdef HAVE_ASYNC_LDS
    __builtin_amdgcn_s_wait_asynccnt(N);
#endif
}

// ---------------------------------------------------------------------------
// Kernel 1: C[M,N] = A[M,K] @ B[K,N], fp32 in/out, bf16 WMMA compute.
// 256 threads = 8 waves, tile BM=128 x BN=128 x BK=32, waves in 4x2 grid,
// each wave owns a 32x64 sub-tile (2x4 WMMA tiles = 8 WMMA per K-step).
// A tiles: async-DMA'd to LDS as fp32 (double buffered), convert at consume.
// B tiles: converted to bf16 at staging, stored transposed [n][k].
// ---------------------------------------------------------------------------
#define BM 128
#define BN 128
#define BK 32

__global__ __launch_bounds__(256) void gemm_bf16_wmma(
    const float* __restrict__ A, const float* __restrict__ Bw,
    float* __restrict__ C, int M, int N, int K)
{
    __shared__ __align__(16) float  sA[2][BM * BK];   // fp32 [m][k]  (2x16KB)
    __shared__ __align__(32) __bf16 sB[2][BN * BK];   // bf16 [n][k]  (2x8KB)

    const int tid  = threadIdx.x;
    const int lane = tid & 31;
    const int wave = tid >> 5;
    const int hi   = lane >> 4;
    const int ln   = lane & 15;
    const int wm   = (wave >> 1) * 32;   // 4 wave-rows cover 128
    const int wn   = (wave & 1) * 64;    // 2 wave-cols cover 128
    const int m0   = blockIdx.y * BM;
    const int n0   = blockIdx.x * BN;

    // per-thread staging coordinates
    const int ar = tid >> 3, ac = (tid & 7) * 4;     // A: 4 rows of stride 32
    const int br = tid >> 5, bc = (tid & 31) * 4;    // B: 4 rows of stride 8

    auto issueA = [&](int buf, int k0) {
        #pragma unroll
        for (int it = 0; it < 4; ++it) {
            int r = ar + it * 32;
            copy16_g2l(A + (size_t)(m0 + r) * K + k0 + ac,
                       &sA[buf][r * BK + ac]);
        }
    };
    auto loadB = [&](int k0, float4* regs) {
        #pragma unroll
        for (int it = 0; it < 4; ++it) {
            int r = br + it * 8;
            regs[it] = *(const float4*)(Bw + (size_t)(k0 + r) * N + n0 + bc);
        }
    };
    auto storeB = [&](int buf, const float4* regs) {
        #pragma unroll
        for (int it = 0; it < 4; ++it) {
            int r = br + it * 8;
            sB[buf][(bc + 0) * BK + r] = tobf(regs[it].x);
            sB[buf][(bc + 1) * BK + r] = tobf(regs[it].y);
            sB[buf][(bc + 2) * BK + r] = tobf(regs[it].z);
            sB[buf][(bc + 3) * BK + r] = tobf(regs[it].w);
        }
    };

    v8f acc[2][4];
    #pragma unroll
    for (int a = 0; a < 2; ++a)
        #pragma unroll
        for (int bI = 0; bI < 4; ++bI) acc[a][bI] = (v8f){};

    const int KT = K / BK;
    float4 bregs[4];
    issueA(0, 0);
    loadB(0, bregs);

    for (int kt = 0; kt < KT; ++kt) {
        const int buf = kt & 1;
        storeB(buf, bregs);
        if (kt + 1 < KT) {
            issueA(buf ^ 1, (kt + 1) * BK);     // overlap with compute below
            loadB((kt + 1) * BK, bregs);
            if (kt + 2 < KT)
                __builtin_prefetch(Bw + (size_t)((kt + 2) * BK + br) * N + n0 + bc, 0, 0);
            wait_async<4>();                    // retire current tile's copies
        } else {
            wait_async<0>();
        }
        __syncthreads();

        // ---- fragments (CDNA5 16-bit A 16x32 / B 32x16 layouts)
        v16bf afrag[2], bfrag[4];
        #pragma unroll
        for (int mt = 0; mt < 2; ++mt) {
            const float* ap = &sA[buf][(wm + mt * 16 + ln) * BK + hi * 16];
            #pragma unroll
            for (int e = 0; e < 8; ++e) {
                afrag[mt][e]     = tobf(ap[e]);       // K = hi*16 + e
                afrag[mt][8 + e] = tobf(ap[8 + e]);   // K = hi*16 + 8 + e
            }
        }
        #pragma unroll
        for (int nt = 0; nt < 4; ++nt)
            bfrag[nt] = *(const v16bf*)&sB[buf][(wn + nt * 16 + ln) * BK + hi * 16];

        #pragma unroll
        for (int mt = 0; mt < 2; ++mt)
            #pragma unroll
            for (int nt = 0; nt < 4; ++nt)
                acc[mt][nt] = __builtin_amdgcn_wmma_f32_16x16x32_bf16(
                    false, afrag[mt], false, bfrag[nt],
                    (short)0, acc[mt][nt], false, false);
        __syncthreads();
    }

    #pragma unroll
    for (int mt = 0; mt < 2; ++mt)
        #pragma unroll
        for (int nt = 0; nt < 4; ++nt) {
            int cc = n0 + wn + nt * 16 + ln;
            #pragma unroll
            for (int i = 0; i < 8; ++i) {
                int rr = m0 + wm + mt * 16 + hi * 8 + i;
                C[(size_t)rr * N + cc] = acc[mt][nt][i];
            }
        }
}

// ---------------------------------------------------------------------------
// Kernel 2: per-head RMSNorm + YaRN RoPE, in place. One wave per (row, head).
// ---------------------------------------------------------------------------
__global__ void norm_rope_kernel(float* __restrict__ X, int ld, int nheads,
                                 const float* __restrict__ w,
                                 const float* __restrict__ invf)
{
    const int row  = blockIdx.x;          // b*T + t
    const int wave = threadIdx.x >> 5;
    const int lane = threadIdx.x & 31;
    if (wave >= nheads) return;

    float* p = X + (size_t)row * ld + wave * DH;
    float e0 = p[lane], e1 = p[lane + 32], e2 = p[lane + 64], e3 = p[lane + 96];

    float ss = e0 * e0 + e1 * e1 + e2 * e2 + e3 * e3;
    #pragma unroll
    for (int m = 16; m >= 1; m >>= 1) ss += __shfl_xor(ss, m, 32);
    float r = rsqrtf(ss * (1.0f / 128.0f) + RMS_EPS);

    e0 *= r * w[lane];      e1 *= r * w[lane + 32];
    e2 *= r * w[lane + 64]; e3 *= r * w[lane + 96];

    float t  = (float)(row % Tt);
    float f0 = t * invf[lane];
    float f1 = t * invf[lane + 32];
    float c0 = cosf(f0), s0 = sinf(f0);
    float c1 = cosf(f1), s1 = sinf(f1);

    p[lane]      = e0 * c0 - e2 * s0;
    p[lane + 64] = e2 * c0 + e0 * s0;
    p[lane + 32] = e1 * c1 - e3 * s1;
    p[lane + 96] = e3 * c1 + e1 * s1;
}

// ---------------------------------------------------------------------------
// Kernel 3: causal flash attention. grid=(T/64, NQ, B), block=128 (4 waves),
// each wave owns 16 q rows. KV tiles of 64 keys staged as bf16 in LDS
// (V transposed): 16 score WMMAs + 16 P@V WMMAs per tile per wave.
// Online softmax stats in registers with the (i + 8*hi) C-layout row mapping.
// ---------------------------------------------------------------------------
#define FT 64

__global__ __launch_bounds__(128) void flash_attn_wmma(
    const float* __restrict__ Q,   // [B*T, NQ*DH] normalized+roped
    const float* __restrict__ KV,  // [B*T, 2*NKV*DH]
    float* __restrict__ O)         // [B*T, NQ*DH]
{
    __shared__ __align__(32) __bf16 sK[FT * DH];      // [key][d]   16KB
    __shared__ __align__(32) __bf16 sVT[DH * FT];     // [d][key]   16KB
    __shared__ __align__(32) __bf16 sP[4][16 * FT];   // per-wave [row][key] 8KB

    const int qblk = blockIdx.x;
    const int h    = blockIdx.y;
    const int b    = blockIdx.z;
    const int g    = h & (NKV - 1);

    const int tid  = threadIdx.x;
    const int lane = tid & 31;
    const int wave = tid >> 5;
    const int hi   = lane >> 4;
    const int ln   = lane & 15;
    const int qt   = qblk * 64 + wave * 16;

    // ---- q fragments once (A layout), ATTN_SCALE folded in
    v16bf qf[4];
    {
        const float* qp0 = Q + ((size_t)(b * Tt + qt + ln)) * (NQ * DH) + h * DH;
        #pragma unroll
        for (int ds = 0; ds < 4; ++ds) {
            const float* qp = qp0 + ds * 32 + hi * 16;
            #pragma unroll
            for (int e = 0; e < 8; ++e) {
                qf[ds][e]     = tobf(qp[e]      * ATTN_SCALE);
                qf[ds][8 + e] = tobf(qp[8 + e] * ATTN_SCALE);
            }
        }
    }

    const float NEG_INF = -__builtin_inff();
    float mrow[8], lrow[8], alpha[8];
    v8f   o[8];
    #pragma unroll
    for (int i = 0; i < 8; ++i) { mrow[i] = NEG_INF; lrow[i] = 0.0f; }
    #pragma unroll
    for (int dt = 0; dt < 8; ++dt) o[dt] = (v8f){};

    const int ntiles = qblk + 1;   // 64-key tiles, causal

    for (int kt = 0; kt < ntiles; ++kt) {
        const int k0 = kt * FT;

        // ---- cooperative K/V staging, fp32 -> bf16, V transposed
        #pragma unroll
        for (int it = 0; it < 16; ++it) {
            int idx = tid + it * 128;
            int r = idx >> 5, c = (idx & 31) * 4;    // r = key 0..63, c = d
            size_t grow = (size_t)(b * Tt + k0 + r) * (2 * NKV * DH);
            float4 kv4 = *(const float4*)(KV + grow + g * DH + c);
            __bf16* kd = &sK[r * DH + c];
            kd[0] = tobf(kv4.x); kd[1] = tobf(kv4.y);
            kd[2] = tobf(kv4.z); kd[3] = tobf(kv4.w);
            float4 vv4 = *(const float4*)(KV + grow + NKV * DH + g * DH + c);
            sVT[(c + 0) * FT + r] = tobf(vv4.x);
            sVT[(c + 1) * FT + r] = tobf(vv4.y);
            sVT[(c + 2) * FT + r] = tobf(vv4.z);
            sVT[(c + 3) * FT + r] = tobf(vv4.w);
        }
        __syncthreads();

        // ---- scores S[16x64] = q @ k^T (four 16-key subtiles)
        v8f s[4];
        #pragma unroll
        for (int nt = 0; nt < 4; ++nt) {
            v8f accs = (v8f){};
            #pragma unroll
            for (int ds = 0; ds < 4; ++ds) {
                v16bf kb = *(const v16bf*)&sK[(nt * 16 + ln) * DH + ds * 32 + hi * 16];
                accs = __builtin_amdgcn_wmma_f32_16x16x32_bf16(
                    false, qf[ds], false, kb, (short)0, accs, false, false);
            }
            s[nt] = accs;
        }

        // ---- causal mask + online softmax (rows = i + 8*hi)
        #pragma unroll
        for (int i = 0; i < 8; ++i) {
            int qr = qt + hi * 8 + i;
            float sv[4];
            #pragma unroll
            for (int nt = 0; nt < 4; ++nt)
                sv[nt] = (k0 + nt * 16 + ln <= qr) ? s[nt][i] : NEG_INF;
            float tmax = fmaxf(fmaxf(sv[0], sv[1]), fmaxf(sv[2], sv[3]));
            tmax = fmaxf(tmax, __shfl_xor(tmax, 1, 32));
            tmax = fmaxf(tmax, __shfl_xor(tmax, 2, 32));
            tmax = fmaxf(tmax, __shfl_xor(tmax, 4, 32));
            tmax = fmaxf(tmax, __shfl_xor(tmax, 8, 32));
            float mnew = fmaxf(mrow[i], tmax);
            alpha[i] = __expf(mrow[i] - mnew);
            mrow[i] = mnew;
            float rs = 0.0f;
            #pragma unroll
            for (int nt = 0; nt < 4; ++nt) {
                float pv = __expf(sv[nt] - mnew);
                rs += pv;
                sP[wave][(hi * 8 + i) * FT + nt * 16 + ln] = tobf(pv);
            }
            rs += __shfl_xor(rs, 1, 32);
            rs += __shfl_xor(rs, 2, 32);
            rs += __shfl_xor(rs, 4, 32);
            rs += __shfl_xor(rs, 8, 32);
            lrow[i] = lrow[i] * alpha[i] + rs;
        }

        #pragma unroll
        for (int dt = 0; dt < 8; ++dt)
            #pragma unroll
            for (int i = 0; i < 8; ++i)
                o[dt][i] *= alpha[i];

        // ---- O += P @ V  (two 32-key K-steps per 64-key tile)
        v16bf pf[2];
        #pragma unroll
        for (int kk = 0; kk < 2; ++kk) {
            const __bf16* pp = &sP[wave][0] + ln * FT + kk * 32 + hi * 16;
            v8bf plo = *(const v8bf*)pp;
            v8bf phi = *(const v8bf*)(pp + 8);
            pf[kk] = __builtin_shufflevector(plo, phi,
                0, 1, 2, 3, 4, 5, 6, 7, 8, 9, 10, 11, 12, 13, 14, 15);
        }
        #pragma unroll
        for (int dt = 0; dt < 8; ++dt) {
            #pragma unroll
            for (int kk = 0; kk < 2; ++kk) {
                v16bf vb = *(const v16bf*)&sVT[(dt * 16 + ln) * FT + kk * 32 + hi * 16];
                o[dt] = __builtin_amdgcn_wmma_f32_16x16x32_bf16(
                    false, pf[kk], false, vb, (short)0, o[dt], false, false);
            }
        }
        __syncthreads();
    }

    #pragma unroll
    for (int dt = 0; dt < 8; ++dt) {
        #pragma unroll
        for (int i = 0; i < 8; ++i) {
            int t = qt + hi * 8 + i;
            O[((size_t)(b * Tt + t)) * (NQ * DH) + h * DH + dt * 16 + ln] =
                o[dt][i] / lrow[i];
        }
    }
}

// ---------------------------------------------------------------------------
// Host launcher
// ---------------------------------------------------------------------------
extern "C" void kernel_launch(void* const* d_in, const int* in_sizes, int n_in,
                              void* d_out, int out_size, void* d_ws, size_t ws_size,
                              hipStream_t stream) {
    const float* x    = (const float*)d_in[0];
    const float* Wq   = (const float*)d_in[1];
    const float* Wkv  = (const float*)d_in[2];
    const float* Wout = (const float*)d_in[3];
    const float* qnw  = (const float*)d_in[4];
    const float* knw  = (const float*)d_in[5];
    const float* invf = (const float*)d_in[6];
    float* out = (float*)d_out;

    const int BT = 2 * Tt;
    float* qbuf  = (float*)d_ws;
    float* kvbuf = qbuf + (size_t)BT * (NQ * DH);
    float* abuf  = kvbuf + (size_t)BT * (2 * NKV * DH);
    (void)ws_size; (void)in_sizes; (void)n_in; (void)out_size;

    gemm_bf16_wmma<<<dim3((NQ * DH) / BN, BT / BM), 256, 0, stream>>>(
        x, Wq, qbuf, BT, NQ * DH, Dd);
    gemm_bf16_wmma<<<dim3((2 * NKV * DH) / BN, BT / BM), 256, 0, stream>>>(
        x, Wkv, kvbuf, BT, 2 * NKV * DH, Dd);
    norm_rope_kernel<<<BT, NQ * 32, 0, stream>>>(qbuf, NQ * DH, NQ, qnw, invf);
    norm_rope_kernel<<<BT, NKV * 32, 0, stream>>>(kvbuf, 2 * NKV * DH, NKV, knw, invf);
    flash_attn_wmma<<<dim3(Tt / 64, NQ, 2), 128, 0, stream>>>(qbuf, kvbuf, abuf);
    gemm_bf16_wmma<<<dim3(Dd / BN, BT / BM), 256, 0, stream>>>(
        abuf, Wout, out, BT, Dd, NQ * DH);
}
